// L3Layer_13735305412631
// MI455X (gfx1250) — matrix operand order repaired and checked by reading, not compiled, not run on hardware
//
#include <hip/hip_runtime.h>
#include <hip/hip_bf16.h>

// ---------------- problem constants ----------------
#define BT      8192        // B*T tokens
#define CDIM    512
#define DUP     2048
#define CATD    2560        // DUP + CDIM
#define KMAX    16
#define NEMB    262144
#define EPS     1e-6f

// ---------------- WMMA vector types ----------------
typedef __attribute__((ext_vector_type(16))) __bf16 v16bf;
typedef __attribute__((ext_vector_type(8)))  __bf16 v8bf;
typedef __attribute__((ext_vector_type(8)))  float  v8f;

// ---------------------------------------------------------------------------
// A-fragment loader: 16x32 bf16 A tile from row-major LDS [16][ldk].
// Per ISA 7.12.2 (16-bit A 16x32): lanes 0-15 -> M=0..15 with K {0..7,16..23},
// lanes 16-31 -> same M with K {8..15,24..31}.
// ---------------------------------------------------------------------------
__device__ __forceinline__ v16bf load_A_lds(const __bf16* base, int ldk, int kbase, int lane) {
    const int m  = lane & 15;
    const int kh = (lane >> 4) << 3;   // 0 or 8
    const __bf16* p = base + m * ldk + kbase + kh;
    v8bf lo = *(const v8bf*)(p);        // K = kh + 0..7
    v8bf hi = *(const v8bf*)(p + 16);   // K = kh + 16..23
    return __builtin_shufflevector(lo, hi, 0,1,2,3,4,5,6,7,8,9,10,11,12,13,14,15);
}

// ---------------------------------------------------------------------------
// B-fragment loader: 32x16 bf16 B tile. Weights stored row-major [N][K] bf16
// so B[k][n] = w[n*ldk + k]. Lanes 0-15 -> N, K {0..7,16..23}; lanes 16-31 -> K {8..15,24..31}.
// ---------------------------------------------------------------------------
__device__ __forceinline__ v16bf load_B_gbl(const __bf16* w, int ldk, int n0, int k0, int lane) {
    const int n  = n0 + (lane & 15);
    const int kb = k0 + ((lane >> 4) << 3);   // +0 or +8
    const __bf16* q = w + (size_t)n * ldk + kb;
    v8bf lo = *(const v8bf*)(q);
    v8bf hi = *(const v8bf*)(q + 16);
    return __builtin_shufflevector(lo, hi, 0,1,2,3,4,5,6,7,8,9,10,11,12,13,14,15);
}

// ===========================================================================
// Kernel 0: one-time f32 -> bf16 conversion of w_up / w_mix into workspace
// ===========================================================================
__global__ void convert_weights_kernel(const float* __restrict__ wup,
                                       const float* __restrict__ wmix,
                                       __bf16* __restrict__ wupBf,
                                       __bf16* __restrict__ wmixBf) {
    const int i = blockIdx.x * blockDim.x + threadIdx.x;
    const int n1 = DUP * CDIM;     // 1048576
    const int n2 = CDIM * CATD;    // 1310720
    if (i < n1) wupBf[i]  = (__bf16)wup[i];
    if (i < n2) wmixBf[i] = (__bf16)wmix[i];
}

// ===========================================================================
// Kernel 1: ragged KV gather + masked softmax attention (one token / block).
// Stages the <=16 KV rows in LDS once; scores and V-sum both read LDS.
// Writes agg as bf16 (GEMM1 A operand precision).
// ===========================================================================
__global__ __launch_bounds__(256) void gather_attn_kernel(
    const float* __restrict__ x,          // [BT, C]
    const int*   __restrict__ token_ids,  // [BT]
    const int*   __restrict__ bounds,     // [VOCAB+1]
    const float* __restrict__ kv,         // [NEMB, C]
    __bf16*      __restrict__ aggBf)      // [BT, C]
{
    __shared__ float xs[CDIM];
    __shared__ float kvs[KMAX * CDIM];
    __shared__ float scores[KMAX];
    __shared__ float ssq;

    const int tid  = threadIdx.x;
    const int lane = tid & 31;
    const int wave = tid >> 5;
    const int n    = blockIdx.x;

    // ---- x row + sum of squares (RMS folded into scores) ----
    const float a0 = x[(size_t)n * CDIM + tid];
    const float a1 = x[(size_t)n * CDIM + 256 + tid];
    xs[tid]       = a0;
    xs[tid + 256] = a1;
    if (tid == 0) ssq = 0.0f;
    float p = a0 * a0 + a1 * a1;
    #pragma unroll
    for (int o = 16; o > 0; o >>= 1) p += __shfl_xor(p, o, 32);
    __syncthreads();
    if (lane == 0) atomicAdd(&ssq, p);

    const int tok   = token_ids[n];
    const int start = bounds[tok];
    const int len   = bounds[tok + 1] - start;
    __syncthreads();
    const float scale = rsqrtf(ssq * (1.0f / (float)CDIM) + EPS);

    // ---- each wave: rows k = wave and wave+8; stage row in LDS + dot ----
    #pragma unroll
    for (int kk = 0; kk < 2; ++kk) {
        const int k = wave + kk * 8;
        int idx = start + k;
        if (idx > NEMB - 1) idx = NEMB - 1;          // clip like the reference
        const float4* srow = (const float4*)(kv + (size_t)idx * CDIM);
        float4*       drow = (float4*)(kvs + k * CDIM);
        float dot = 0.0f;
        #pragma unroll
        for (int j = 0; j < 4; ++j) {
            const int vi = lane + 32 * j;            // coalesced float4 index
            const float4 v  = srow[vi];
            drow[vi] = v;
            const float4 xv = *(const float4*)(xs + vi * 4);
            dot += v.x * xv.x + v.y * xv.y + v.z * xv.z + v.w * xv.w;
        }
        #pragma unroll
        for (int o = 16; o > 0; o >>= 1) dot += __shfl_xor(dot, o, 32);
        if (lane == 0)
            scores[k] = (k < len) ? dot * scale : -__builtin_inff();
    }
    __syncthreads();

    // ---- masked, stable softmax (redundant per thread, 16 scalars) ----
    float m = -__builtin_inff();
    float s[KMAX];
    #pragma unroll
    for (int k = 0; k < KMAX; ++k) { s[k] = scores[k]; m = fmaxf(m, s[k]); }
    if (!(m > -__builtin_inff())) m = 0.0f;          // fully-masked row guard
    float e[KMAX];
    float denom = 0.0f;
    #pragma unroll
    for (int k = 0; k < KMAX; ++k) {
        e[k] = (k < len) ? __expf(s[k] - m) : 0.0f;
        denom += e[k];
    }
    const float inv = (denom > 0.0f) ? (1.0f / denom) : 0.0f;

    // ---- weighted V sum from LDS; write bf16 agg ----
    float acc0 = 0.0f, acc1 = 0.0f;
    #pragma unroll
    for (int k = 0; k < KMAX; ++k) {
        const float w = e[k] * inv;
        acc0 += w * kvs[k * CDIM + tid];
        acc1 += w * kvs[k * CDIM + 256 + tid];
    }
    aggBf[(size_t)n * CDIM + tid]       = (__bf16)acc0;
    aggBf[(size_t)n * CDIM + 256 + tid] = (__bf16)acc1;
}

// ===========================================================================
// Kernel 2: fused  up = rmsnorm(agg @ w_up^T);  delta = [up, x] @ w_mix^T
// 16 token-rows per block, 8 waves; all matrix math on v_wmma bf16.
// LDS: up[16][2048] bf16 (64K) | aux[16][512] bf16 (agg then x, 16K) | stats.
// ===========================================================================
__global__ __launch_bounds__(256) void fused_mlp_kernel(
    const __bf16* __restrict__ aggBf,   // [BT, C]
    const float*  __restrict__ x,       // [BT, C]
    const __bf16* __restrict__ wupBf,   // [DUP, C]
    const __bf16* __restrict__ wmixBf,  // [C, CATD]
    float*        __restrict__ out)     // [BT, C]
{
    __shared__ __align__(16) unsigned char smem[16 * DUP * 2 + 16 * CDIM * 2 + 64 + 64];
    __bf16* upL    = (__bf16*)smem;                                   // [16][2048]
    __bf16* auxL   = (__bf16*)(smem + 16 * DUP * 2);                  // [16][512]
    float*  sumsqL = (float*)(smem + 16 * DUP * 2 + 16 * CDIM * 2);   // [16]
    float*  scaleL = sumsqL + 16;                                     // [16]

    const int tid  = threadIdx.x;
    const int lane = tid & 31;
    const int wave = tid >> 5;
    const int hi   = lane >> 4;          // C/D row-half select
    const int col  = lane & 15;          // C/D column within tile
    const int row0 = blockIdx.x * 16;    // token base

    // ---- stage agg rows (already bf16) into LDS ----
    {
        const __bf16* src = aggBf + (size_t)row0 * CDIM;
        for (int i = tid; i < 16 * CDIM; i += 256) auxL[i] = src[i];
        if (tid < 16) sumsqL[tid] = 0.0f;
    }
    __syncthreads();

    // ---- GEMM1: up_raw = agg @ w_up^T ; park raw bf16 in LDS, track sumsq ----
    float rss[8];
    #pragma unroll
    for (int r = 0; r < 8; ++r) rss[r] = 0.0f;

    for (int t = 0; t < 16; ++t) {                     // 16 N-tiles / wave
        const int n0 = wave * 256 + t * 16;
        v8f acc = {};
        #pragma unroll
        for (int ks = 0; ks < CDIM / 32; ++ks) {       // K = 512
            v16bf a = load_A_lds(auxL, CDIM, ks * 32, lane);
            v16bf b = load_B_gbl(wupBf, CDIM, n0, ks * 32, lane);
            acc = __builtin_amdgcn_wmma_f32_16x16x32_bf16(
                      false, a, false, b, (short)0, acc, false, false);
        }
        #pragma unroll
        for (int r = 0; r < 8; ++r) {
            const float v = acc[r];
            rss[r] += v * v;
            upL[(r + hi * 8) * DUP + n0 + col] = (__bf16)v;
        }
    }

    // ---- row sum-of-squares: shfl-reduce within 16-lane halves, then ds atomics ----
    #pragma unroll
    for (int r = 0; r < 8; ++r) {
        float v = rss[r];
        #pragma unroll
        for (int o = 1; o < 16; o <<= 1) v += __shfl_xor(v, o, 32);
        if ((lane & 15) == 0) atomicAdd(&sumsqL[r + hi * 8], v);
    }
    __syncthreads();

    if (tid < 16) scaleL[tid] = rsqrtf(sumsqL[tid] * (1.0f / (float)DUP) + EPS);
    __syncthreads();

    // ---- load x rows (bf16) over agg region; rescale up in place ----
    {
        const float* xsrc = x + (size_t)row0 * CDIM;
        for (int i = tid; i < 16 * CDIM; i += 256) auxL[i] = (__bf16)xsrc[i];
        for (int i = tid; i < 16 * DUP; i += 256) {
            const int r = i >> 11;
            upL[i] = (__bf16)((float)upL[i] * scaleL[r]);
        }
    }
    __syncthreads();

    // ---- GEMM2: delta = [up, x] @ w_mix^T  (K = 2560) ----
    for (int t = 0; t < 4; ++t) {                      // 4 N-tiles / wave -> 512 cols
        const int n0 = wave * 64 + t * 16;
        v8f acc = {};
        for (int ks = 0; ks < DUP / 32; ++ks) {        // first 2048 K from up
            v16bf a = load_A_lds(upL, DUP, ks * 32, lane);
            v16bf b = load_B_gbl(wmixBf, CATD, n0, ks * 32, lane);
            acc = __builtin_amdgcn_wmma_f32_16x16x32_bf16(
                      false, a, false, b, (short)0, acc, false, false);
        }
        #pragma unroll
        for (int ks = 0; ks < CDIM / 32; ++ks) {       // last 512 K from x
            v16bf a = load_A_lds(auxL, CDIM, ks * 32, lane);
            v16bf b = load_B_gbl(wmixBf, CATD, n0, DUP + ks * 32, lane);
            acc = __builtin_amdgcn_wmma_f32_16x16x32_bf16(
                      false, a, false, b, (short)0, acc, false, false);
        }
        #pragma unroll
        for (int r = 0; r < 8; ++r) {
            const int row = r + hi * 8;
            out[(size_t)(row0 + row) * CDIM + n0 + col] = acc[r];
        }
    }
}

// ===========================================================================
// Host launcher
// ===========================================================================
extern "C" void kernel_launch(void* const* d_in, const int* in_sizes, int n_in,
                              void* d_out, int out_size, void* d_ws, size_t ws_size,
                              hipStream_t stream) {
    const float* x         = (const float*)d_in[0];   // [4,2048,512]
    const int*   token_ids = (const int*)  d_in[1];   // [4,2048]
    const int*   bounds    = (const int*)  d_in[2];   // [16385]
    const float* kv        = (const float*)d_in[3];   // [262144,512]
    const float* wup       = (const float*)d_in[4];   // [2048,512]
    const float* wmix      = (const float*)d_in[5];   // [512,2560]
    float*       out       = (float*)d_out;           // [4,2048,512]

    char* ws = (char*)d_ws;
    __bf16* aggBf  = (__bf16*)(ws);                              // 8,388,608 B
    __bf16* wupBf  = (__bf16*)(ws + (size_t)BT * CDIM * 2);      // 2,097,152 B
    __bf16* wmixBf = (__bf16*)(ws + (size_t)BT * CDIM * 2 + (size_t)DUP * CDIM * 2);

    // 0) weights -> bf16 (once per launch; stays hot in L2)
    {
        const int n = CDIM * CATD;   // max(n1, n2)
        convert_weights_kernel<<<(n + 255) / 256, 256, 0, stream>>>(wup, wmix, wupBf, wmixBf);
    }
    // 1) ragged gather + softmax attention -> agg (bf16)
    gather_attn_kernel<<<BT, 256, 0, stream>>>(x, token_ids, bounds, kv, aggBf);
    // 2) fused GEMM1 + rmsnorm + concat + GEMM2 on WMMA
    fused_mlp_kernel<<<BT / 16, 256, 0, stream>>>(aggBf, x, wupBf, wmixBf, out);
}
